// Attn_62122406969691
// MI455X (gfx1250) — matrix operand order repaired and checked by reading, not compiled, not run on hardware
//
#include <hip/hip_runtime.h>
#include <hip/hip_bf16.h>

#define N_Q   1024
#define M_K   1024
#define ENC   512
#define ATTN  256

typedef _Float16 v16h __attribute__((ext_vector_type(16)));
typedef _Float16 v8h  __attribute__((ext_vector_type(8)));
typedef float    v8f  __attribute__((ext_vector_type(8)));

// ---------------- fast transcendental helpers (CDNA5 TRANS32 ops) ----------
__device__ __forceinline__ float fast_tanh(float x) {
#if __has_builtin(__builtin_amdgcn_tanhf)
    return __builtin_amdgcn_tanhf(x);          // v_tanh_f32
#else
    return tanhf(x);
#endif
}
__device__ __forceinline__ float fast_exp2(float x) {
#if __has_builtin(__builtin_amdgcn_exp2f)
    return __builtin_amdgcn_exp2f(x);          // v_exp_f32
#else
    return exp2f(x);
#endif
}
__device__ __forceinline__ float fast_rcp(float x) {
#if __has_builtin(__builtin_amdgcn_rcpf)
    return __builtin_amdgcn_rcpf(x);           // v_rcp_f32
#else
    return 1.0f / x;
#endif
}
__device__ __forceinline__ void sched_fence() {
#if __has_builtin(__builtin_amdgcn_sched_barrier)
    __builtin_amdgcn_sched_barrier(0);
#else
    asm volatile("" ::: "memory");
#endif
}

__device__ __forceinline__ v16h cat16(v8h lo, v8h hi) {
    return __builtin_shufflevector(lo, hi, 0,1,2,3,4,5,6,7,8,9,10,11,12,13,14,15);
}

// ---------------------------------------------------------------------------
// Elementwise f32 -> f16 convert (8 elems/thread). Done once so the GEMM inner
// loops carry no conversions and half the register traffic.
// ---------------------------------------------------------------------------
__global__ __launch_bounds__(256)
void cvt_f32_f16(const float* __restrict__ in, _Float16* __restrict__ out)
{
    const int i = blockIdx.x * 256 + threadIdx.x;
    const float4* p = (const float4*)in + (size_t)i * 2;
    const float4 x0 = p[0], x1 = p[1];
    v8h r;
    r[0] = (_Float16)x0.x; r[1] = (_Float16)x0.y; r[2] = (_Float16)x0.z; r[3] = (_Float16)x0.w;
    r[4] = (_Float16)x1.x; r[5] = (_Float16)x1.y; r[6] = (_Float16)x1.z; r[7] = (_Float16)x1.w;
    *((v8h*)out + i) = r;
}

// ---------------------------------------------------------------------------
// Unified f16 GEMM: C[1024, 256] = A[1024, K] @ BT[256, K]^T (both row-major
// with K contiguous). One wave computes a 16x64 tile, 4 WMMAs per 32-K step,
// double-buffered: stage k+1's 10 global b128 loads are issued (and pinned by
// a sched_barrier) before stage k's WMMAs, so loads overlap matrix math.
// Fragment lane layouts (this code's own k order, shared by A and B):
//   A: lane (rr=l&15, h=l>>4): row rr, K = h*16..h*16+15 (contiguous)
//   B: lane: col rr, K = h*16..h*16+15 (contiguous)
// MODE 0: f32 out + bias.  MODE 1: f16 TRANSPOSED out + bias.  MODE 2: f32 out.
// ---------------------------------------------------------------------------
struct Stage {
    v8h al, ah;
    v8h bl[4], bh[4];
};

template<int K>
__device__ __forceinline__ void load_stage(Stage& s, const _Float16* arow,
                                           const _Float16* const brow[4],
                                           int k0, int h)
{
    s.al = *(const v8h*)(arow + k0 + h * 16);
    s.ah = *(const v8h*)(arow + k0 + h * 16 + 8);
#pragma unroll
    for (int f = 0; f < 4; ++f) {
        s.bl[f] = *(const v8h*)(brow[f] + k0);
        s.bh[f] = *(const v8h*)(brow[f] + k0 + 8);
    }
}

__device__ __forceinline__ void mm_step(const Stage& s, v8f c[4])
{
    const v16h a = cat16(s.al, s.ah);
#pragma unroll
    for (int f = 0; f < 4; ++f)
        c[f] = __builtin_amdgcn_wmma_f32_16x16x32_f16(
                   false, a, false, cat16(s.bl[f], s.bh[f]), (short)0, c[f], false, false);
}

template<int K, int MODE>
__global__ __launch_bounds__(32)
void gemm16(const _Float16* __restrict__ A,    // [1024][K]
            const _Float16* __restrict__ BT,   // [256][K]
            const float* __restrict__ bias,
            float* __restrict__ outF32, _Float16* __restrict__ outT16)
{
    const int n0 = (blockIdx.x >> 2) * 16;   // 64 row tiles
    const int og = (blockIdx.x & 3) * 64;    // 4 col groups of 64
    const int l  = threadIdx.x;
    const int rr = l & 15;
    const int h  = l >> 4;
    const int kb = h * 16;

    const _Float16* arow = A + (size_t)(n0 + rr) * K;
    const _Float16* brow[4];
#pragma unroll
    for (int f = 0; f < 4; ++f)
        brow[f] = BT + (size_t)(og + f * 16 + rr) * K + kb;

    v8f c[4] = {};
    Stage s0, s1;
    load_stage<K>(s0, arow, brow, 0, h);

    for (int k0 = 0; k0 < K; k0 += 64) {
        load_stage<K>(s1, arow, brow, k0 + 32, h);   // prefetch stage k+1
        sched_fence();                               // keep loads above WMMAs
        mm_step(s0, c);

        const int kn = (k0 + 64 < K) ? (k0 + 64) : 0;  // benign reload on last iter
        load_stage<K>(s0, arow, brow, kn, h);
        sched_fence();
        mm_step(s1, c);
    }

    // D layout: VGPR i, lanes 0-15 -> (M=i, N=lane); lanes 16-31 -> (M=8+i)
#pragma unroll
    for (int f = 0; f < 4; ++f) {
        const int col = og + f * 16 + rr;
        if (MODE == 0) {
            const float bv = bias[col];
#pragma unroll
            for (int i = 0; i < 8; ++i)
                outF32[(size_t)(n0 + 8 * h + i) * ATTN + col] = c[f][i] + bv;
        } else if (MODE == 1) {
            const float bv = bias[col];
            v8h p;
#pragma unroll
            for (int i = 0; i < 8; ++i) p[i] = (_Float16)(c[f][i] + bv);
            *(v8h*)(outT16 + (size_t)col * M_K + n0 + 8 * h) = p;  // transposed
        } else {
#pragma unroll
            for (int i = 0; i < 8; ++i)
                outF32[(size_t)(n0 + 8 * h + i) * ATTN + col] = c[f][i];
        }
    }
}

// ---------------------------------------------------------------------------
// Fused additive scores + row softmax.  THE hot kernel: 268M v_tanh_f32.
// Block = 256 threads handles 8 query rows; thread t owns keys m = t + i*256.
// Each kp float4 load feeds 8 rows x 4 lanes = 32 tanh+FMA; qs/Ww served as
// ds_load_b128 broadcasts; reductions are wave32 shuffles + 8-entry LDS combine.
// ---------------------------------------------------------------------------
__global__ __launch_bounds__(256)
void score_softmax(const float* __restrict__ qp, const float* __restrict__ kp,
                   const float* __restrict__ Ww, const float* __restrict__ Wb,
                   const unsigned char* __restrict__ mask,
                   _Float16* __restrict__ wgt)
{
    const int n0 = blockIdx.x * 8;
    const int t  = threadIdx.x;

    __shared__ __align__(16) float qs[8][ATTN];
    __shared__ __align__(16) float wsh[ATTN];
    __shared__ float red8[8];

    wsh[t] = Ww[t];
#pragma unroll
    for (int r = 0; r < 8; ++r) qs[r][t] = qp[(size_t)(n0 + r) * ATTN + t];
    __syncthreads();

    float acc[8][4];
#pragma unroll
    for (int r = 0; r < 8; ++r)
#pragma unroll
        for (int i = 0; i < 4; ++i) acc[r][i] = 0.0f;

    for (int a = 0; a < ATTN; a += 4) {
        const float4 wv = *(const float4*)&wsh[a];
        float4 qv[8];
#pragma unroll
        for (int r = 0; r < 8; ++r) qv[r] = *(const float4*)&qs[r][a];
#pragma unroll
        for (int i = 0; i < 4; ++i) {
            const float4 kv = *(const float4*)(kp + (size_t)(t + i * 256) * ATTN + a);
#pragma unroll
            for (int r = 0; r < 8; ++r) {
                acc[r][i] += wv.x * fast_tanh(qv[r].x + kv.x)
                           + wv.y * fast_tanh(qv[r].y + kv.y)
                           + wv.z * fast_tanh(qv[r].z + kv.z)
                           + wv.w * fast_tanh(qv[r].w + kv.w);
            }
        }
    }

    const float wb = Wb[0];
#pragma unroll
    for (int r = 0; r < 8; ++r)
#pragma unroll
        for (int i = 0; i < 4; ++i) {
            const float s = acc[r][i] + wb;
            acc[r][i] = mask[(size_t)(n0 + r) * M_K + t + i * 256] ? s : -1.0e6f;
        }

    const int wid = t >> 5, lid = t & 31;
#pragma unroll
    for (int r = 0; r < 8; ++r) {
        float lm = fmaxf(fmaxf(acc[r][0], acc[r][1]), fmaxf(acc[r][2], acc[r][3]));
#pragma unroll
        for (int off = 16; off > 0; off >>= 1)
            lm = fmaxf(lm, __shfl_xor(lm, off, 32));
        if (lid == 0) red8[wid] = lm;
        __syncthreads();
        float mx = red8[0];
#pragma unroll
        for (int w = 1; w < 8; ++w) mx = fmaxf(mx, red8[w]);
        __syncthreads();

        float e[4];
        float ls = 0.0f;
#pragma unroll
        for (int i = 0; i < 4; ++i) {
            e[i] = fast_exp2((acc[r][i] - mx) * 1.44269504089f);
            ls += e[i];
        }
#pragma unroll
        for (int off = 16; off > 0; off >>= 1)
            ls += __shfl_xor(ls, off, 32);
        if (lid == 0) red8[wid] = ls;
        __syncthreads();
        float tot = red8[0];
#pragma unroll
        for (int w = 1; w < 8; ++w) tot += red8[w];
        const float inv = fast_rcp(tot);
#pragma unroll
        for (int i = 0; i < 4; ++i)
            wgt[(size_t)(n0 + r) * M_K + t + i * 256] = (_Float16)(e[i] * inv);
        __syncthreads();   // red8 reused next row
    }
}

// ---------------------------------------------------------------------------
extern "C" void kernel_launch(void* const* d_in, const int* in_sizes, int n_in,
                              void* d_out, int out_size, void* d_ws, size_t ws_size,
                              hipStream_t stream)
{
    (void)in_sizes; (void)n_in; (void)out_size; (void)ws_size;

    const float* q  = (const float*)d_in[0];
    const float* k  = (const float*)d_in[1];
    const float* v  = (const float*)d_in[2];
    const unsigned char* mask = (const unsigned char*)d_in[3];
    const float* Qw = (const float*)d_in[4];
    const float* Qb = (const float*)d_in[5];
    const float* Kw = (const float*)d_in[6];
    const float* Kb = (const float*)d_in[7];
    const float* Vw = (const float*)d_in[8];
    const float* Vb = (const float*)d_in[9];
    const float* Ww = (const float*)d_in[10];
    const float* Wb = (const float*)d_in[11];
    float* out = (float*)d_out;

    char* ws = (char*)d_ws;
    float*    qp  = (float*)(ws);                                // 1 MB
    float*    kp  = (float*)(ws + (1u << 20));                   // 1 MB
    _Float16* vpT = (_Float16*)(ws + (2u << 20));                // 0.5 MB (transposed)
    _Float16* wgt = (_Float16*)(ws + (2u << 20) + (1u << 19));   // 2 MB
    _Float16* qh  = (_Float16*)(ws + (4u << 20) + (1u << 19));   // 1 MB
    _Float16* kh  = (_Float16*)(ws + (5u << 20) + (1u << 19));   // 1 MB
    _Float16* vh  = (_Float16*)(ws + (6u << 20) + (1u << 19));   // 1 MB
    _Float16* Qwh = (_Float16*)(ws + (7u << 20) + (1u << 19));   // 256 KB
    _Float16* Kwh = (_Float16*)(ws + (7u << 20) + (3u << 19));   // 256 KB
    _Float16* Vwh = (_Float16*)(ws + (8u << 20));                // 256 KB

    // one-time f32 -> f16 conversions (8 elems/thread, 2048 elems/block)
    cvt_f32_f16<<<(N_Q * ENC) / 2048, 256, 0, stream>>>(q, qh);
    cvt_f32_f16<<<(M_K * ENC) / 2048, 256, 0, stream>>>(k, kh);
    cvt_f32_f16<<<(M_K * ENC) / 2048, 256, 0, stream>>>(v, vh);
    cvt_f32_f16<<<(ATTN * ENC) / 2048, 256, 0, stream>>>(Qw, Qwh);
    cvt_f32_f16<<<(ATTN * ENC) / 2048, 256, 0, stream>>>(Kw, Kwh);
    cvt_f32_f16<<<(ATTN * ENC) / 2048, 256, 0, stream>>>(Vw, Vwh);

    // projections: 64 row-tiles x 4 col-groups = 256 waves each
    gemm16<ENC, 0><<<256, 32, 0, stream>>>(qh, Qwh, Qb, qp, nullptr);
    gemm16<ENC, 0><<<256, 32, 0, stream>>>(kh, Kwh, Kb, kp, nullptr);
    gemm16<ENC, 1><<<256, 32, 0, stream>>>(vh, Vwh, Vb, nullptr, vpT);

    score_softmax<<<128, 256, 0, stream>>>(qp, kp, Ww, Wb, mask, wgt);

    gemm16<M_K, 2><<<256, 32, 0, stream>>>(wgt, vpT, nullptr, out, nullptr);
}
